// Road2RegionAtten_63393717289654
// MI455X (gfx1250) — matrix operand order repaired and checked by reading, not compiled, not run on hardware
//
#include <hip/hip_runtime.h>
#include <hip/hip_bf16.h>

typedef __attribute__((ext_vector_type(16))) _Float16 v16h;
typedef __attribute__((ext_vector_type(8)))  _Float16 v8h;
typedef __attribute__((ext_vector_type(8)))  float    v8f;

constexpr int NROADS = 50000;
constexpr int RREG   = 512;
constexpr int HDIM   = 512;
constexpr int EDIM   = 64;
constexpr int TILE_N = 32;
constexpr int NSPLIT = 16;
constexpr int SPLIT_LEN = 3136;           // 16*3136 = 50176 >= 50000, divisible by 32
constexpr float INV_SQRT_H = 0.04419417382415922f;  // 1/sqrt(512)

__device__ inline v8h cvt8(v8f x) { return __builtin_convertvector(x, v8h); }

__device__ inline v16h cat16(v8h lo, v8h hi) {
  return __builtin_shufflevector(lo, hi, 0,1,2,3,4,5,6,7,8,9,10,11,12,13,14,15);
}

__device__ inline v8f wmma_f16(v16h a, v16h b, v8f c) {
  return __builtin_amdgcn_wmma_f32_16x16x32_f16(false, a, false, b, (short)0, c, false, false);
}

// B operand from f16 row-major weight row: K = kb + (lane>>4)*8 + e (e<8), kb + 16 + (lane>>4)*8 + (e-8) (e>=8)
__device__ inline v16h load_b_f16(const _Float16* wrow, int kb, int half) {
  const _Float16* p = wrow + kb + half * 8;
  return cat16(*(const v8h*)p, *(const v8h*)(p + 16));
}

// ---------------------------------------------------------------------------
// Bulk f32 -> f16 conversion (for loop-invariant weight matrices).
// ---------------------------------------------------------------------------
__global__ __launch_bounds__(256) void cvt_f16_kernel(
    const float* __restrict__ src, _Float16* __restrict__ dst) {
  int idx = blockIdx.x * 256 + threadIdx.x;  // one v8 chunk each
  v8f x = *(const v8f*)(src + (size_t)idx * 8);
  *(v8h*)(dst + (size_t)idx * 8) = cvt8(x);
}

// ---------------------------------------------------------------------------
// Fused K/V projection: stages 16-road A tile in LDS (f32->f16) once, then
// each wave computes 16x32 of BOTH k and v (4 independent WMMA chains/step).
// grid: (N/16, HDIM/256), block 256 (8 waves); wave w -> cols by*256+w*32.
// Weights pre-converted to f16 (no cvt in the hot loop).
// ---------------------------------------------------------------------------
__global__ __launch_bounds__(256) void gemm_kv_kernel(
    const float* __restrict__ A, const _Float16* __restrict__ Wk_,
    const _Float16* __restrict__ Wv_, _Float16* __restrict__ Ko,
    _Float16* __restrict__ Vo) {
  __shared__ __align__(16) _Float16 s_a[16][HDIM];  // 16 KB
  const int tid = threadIdx.x, wave = tid >> 5, lane = tid & 31;
  const int m_ = lane & 15, half = lane >> 4;
  const int r0 = blockIdx.x * 16;
  const int i0 = blockIdx.y * 256 + wave * 32;

  // stage + convert A tile (16 x 512 f32 -> f16)
  for (int idx = tid; idx < 16 * (HDIM / 8); idx += 256) {
    v8f x = *(const v8f*)(A + (size_t)r0 * HDIM + (size_t)idx * 8);
    ((v8h*)&s_a[0][0])[idx] = cvt8(x);
  }
  __syncthreads();

  const _Float16* wk0 = Wk_ + (size_t)(i0 + m_) * HDIM;
  const _Float16* wk1 = Wk_ + (size_t)(i0 + 16 + m_) * HDIM;
  const _Float16* wv0 = Wv_ + (size_t)(i0 + m_) * HDIM;
  const _Float16* wv1 = Wv_ + (size_t)(i0 + 16 + m_) * HDIM;
  v8f ak0 = {}, ak1 = {}, av0 = {}, av1 = {};
#pragma unroll 4
  for (int kb = 0; kb < HDIM; kb += 32) {
    const _Float16* ar = &s_a[m_][0];
    v16h a = cat16(*(const v8h*)(ar + kb + half * 8),
                   *(const v8h*)(ar + kb + 16 + half * 8));
    ak0 = wmma_f16(a, load_b_f16(wk0, kb, half), ak0);
    ak1 = wmma_f16(a, load_b_f16(wk1, kb, half), ak1);
    av0 = wmma_f16(a, load_b_f16(wv0, kb, half), av0);
    av1 = wmma_f16(a, load_b_f16(wv1, kb, half), av1);
  }
#pragma unroll
  for (int j = 0; j < 8; ++j) {
    size_t row = (size_t)(r0 + j + 8 * half) * HDIM;
    Ko[row + i0 + m_]      = (_Float16)ak0[j];
    Ko[row + i0 + 16 + m_] = (_Float16)ak1[j];
    Vo[row + i0 + m_]      = (_Float16)av0[j];
    Vo[row + i0 + 16 + m_] = (_Float16)av1[j];
  }
}

// ---------------------------------------------------------------------------
// Single-output GEMM (used for q): C[m,i] = (f16) sum_h A[m,h]*W[i,h]
// grid: (M/16, HDIM/128); two-accumulator K-split; f16 weights.
// ---------------------------------------------------------------------------
__global__ __launch_bounds__(256) void gemm_xwT_f16out(
    const float* __restrict__ A, const _Float16* __restrict__ W,
    _Float16* __restrict__ C) {
  const int tid = threadIdx.x, wave = tid >> 5, lane = tid & 31;
  const int m_ = lane & 15, half = lane >> 4;
  const int r0 = blockIdx.x * 16;
  const int i0 = blockIdx.y * 128 + wave * 16;
  const float* arow = A + (size_t)(r0 + m_) * HDIM;
  const _Float16* wrow = W + (size_t)(i0 + m_) * HDIM;
  v8f acc0 = {}, acc1 = {};
#pragma unroll 2
  for (int kb = 0; kb < HDIM; kb += 64) {
    v16h a0 = cat16(cvt8(*(const v8f*)(arow + kb + half * 8)),
                    cvt8(*(const v8f*)(arow + kb + 16 + half * 8)));
    acc0 = wmma_f16(a0, load_b_f16(wrow, kb, half), acc0);
    v16h a1 = cat16(cvt8(*(const v8f*)(arow + kb + 32 + half * 8)),
                    cvt8(*(const v8f*)(arow + kb + 48 + half * 8)));
    acc1 = wmma_f16(a1, load_b_f16(wrow, kb + 32, half), acc1);
  }
#pragma unroll
  for (int j = 0; j < 8; ++j)
    C[(size_t)(r0 + j + 8 * half) * HDIM + i0 + m_] = (_Float16)(acc0[j] + acc1[j]);
}

// ---------------------------------------------------------------------------
// deg_proj[d,r] = deg_table[d,:] . w1[r,:]; len_proj[d,r] (row 0 zeroed)
// ---------------------------------------------------------------------------
__global__ __launch_bounds__(256) void table_proj_kernel(
    const float* __restrict__ deg_table, const float* __restrict__ w1,
    const float* __restrict__ len_table, const float* __restrict__ w2,
    float* __restrict__ deg_proj, float* __restrict__ len_proj) {
  int idx = blockIdx.x * 256 + threadIdx.x;
  if (idx < 360 * RREG) {
    int d = idx >> 9, r = idx & 511;
    float s = 0.f;
    for (int e = 0; e < EDIM; ++e) s += deg_table[d * EDIM + e] * w1[r * EDIM + e];
    deg_proj[idx] = s;
  } else {
    int i2 = idx - 360 * RREG;
    if (i2 < 200 * RREG) {
      int d = i2 >> 9, r = i2 & 511;
      float s = 0.f;
      if (d != 0)
        for (int e = 0; e < EDIM; ++e) s += len_table[d * EDIM + e] * w2[r * EDIM + e];
      len_proj[i2] = s;
    }
  }
}

// ---------------------------------------------------------------------------
// Flash-style masked attention with fused gather scores.
// grid: (R/16, NSPLIT); block 256 (8 waves).  Wave w owns H cols [w*64,+64).
// Per 32-road tile: K-split QK^T via WMMA + ds_add_f32 reduce, fused
// part1/part2/mask gathers, online softmax, P@V via WMMA.
// ---------------------------------------------------------------------------
__global__ __launch_bounds__(256) void flash_attn_kernel(
    const _Float16* __restrict__ q_h, const _Float16* __restrict__ k_h,
    const _Float16* __restrict__ v_h,
    const float* __restrict__ deg_proj, const float* __restrict__ len_proj,
    const int* __restrict__ assign, const int* __restrict__ degm,
    const int* __restrict__ distm,
    float* __restrict__ pO, float* __restrict__ pM, float* __restrict__ pL) {
  __shared__ __align__(16) _Float16 s_q[16][HDIM];       // 16 KB
  __shared__ __align__(16) _Float16 s_vT[HDIM][TILE_N];  // 32 KB (V transposed)
  __shared__ __align__(16) float    s_s[16][TILE_N];     // 2 KB score tile
  __shared__ __align__(16) _Float16 s_p[16][TILE_N];     // 1 KB P (f16)
  __shared__ float s_m[16], s_l[16], s_scale[16];

  const int tid = threadIdx.x, wave = tid >> 5, lane = tid & 31;
  const int m_ = lane & 15, half = lane >> 4;
  const int r0 = blockIdx.x * 16;
  const int split = blockIdx.y;
  const int n_begin = split * SPLIT_LEN;
  const int n_end = min(n_begin + SPLIT_LEN, NROADS);
  const int kk = wave * 64;  // this wave's K(h)-slice for QK^T

  // async preload of 16 q rows: contiguous global -> LDS (ASYNCcnt path)
  {
    const _Float16* qbase = q_h + (size_t)r0 * HDIM;
#pragma unroll
    for (int it = 0; it < 4; ++it) {
      int idx = tid + it * 256;  // 1024 x 16B chunks
      unsigned int lds = (unsigned int)(size_t)(&s_q[0][0]) + idx * 16;
      unsigned long long g = (unsigned long long)(size_t)(qbase + idx * 8);
      asm volatile("global_load_async_to_lds_b128 %0, %1, off"
                   :: "v"(lds), "v"(g) : "memory");
    }
    asm volatile("s_wait_asynccnt 0x0" ::: "memory");
  }
  if (tid < 16) { s_m[tid] = -1e30f; s_l[tid] = 0.f; }

  v8f acc[4] = {{}, {}, {}, {}};
  __syncthreads();

  for (int n0 = n_begin; n0 < n_end; n0 += TILE_N) {
    __syncthreads();  // previous tile fully consumed

    // stage V tile transposed: s_vT[h][road]
    for (int idx = tid; idx < TILE_N * (HDIM / 8); idx += 256) {
      int road = idx >> 6;           // /(HDIM/8)
      int hb = (idx & 63) * 8;
      int g = n0 + road;
      v8h d = {};
      if (g < n_end) d = *(const v8h*)(v_h + (size_t)g * HDIM + hb);
#pragma unroll
      for (int j = 0; j < 8; ++j) s_vT[hb + j][road] = d[j];
    }
    for (int idx = tid; idx < 16 * TILE_N; idx += 256) (&s_s[0][0])[idx] = 0.f;
    __syncthreads();

    // partial S = q_slice @ k_slice^T (K-split across 8 waves)
    if (n0 + TILE_N < n_end)
      __builtin_prefetch(k_h + (size_t)(n0 + TILE_N + m_) * HDIM + kk, 0, 0);
#pragma unroll
    for (int t = 0; t < 2; ++t) {
      v8f c = {};
#pragma unroll
      for (int k2 = 0; k2 < 64; k2 += 32) {
        const int kb = kk + k2;
        const _Float16* qr = &s_q[m_][0];
        v16h a = cat16(*(const v8h*)(qr + kb + half * 8),
                       *(const v8h*)(qr + kb + 16 + half * 8));
        v16h b = {};
        int road = n0 + t * 16 + m_;
        if (road < n_end) {
          const _Float16* kp = k_h + (size_t)road * HDIM + kb + half * 8;
          b = cat16(*(const v8h*)kp, *(const v8h*)(kp + 16));
        }
        c = wmma_f16(a, b, c);
      }
#pragma unroll
      for (int j = 0; j < 8; ++j)
        atomicAdd(&s_s[j + 8 * half][t * 16 + m_], c[j]);
    }
    __syncthreads();

    // fuse part1/part2/mask; adjacent lanes read adjacent regions (coalesced)
    for (int idx = tid; idx < 16 * TILE_N; idx += 256) {
      int row = idx & 15, col = idx >> 4;
      int r = r0 + row, g = n0 + col;
      float s = -1e30f;
      if (g < n_end) {
        size_t off = (size_t)g * RREG + r;
        if (assign[off] == 1) {
          int dg = degm[off], dd = distm[off];
          s = (s_s[row][col] + deg_proj[(size_t)dg * RREG + r] +
               len_proj[(size_t)dd * RREG + r]) * INV_SQRT_H;
        }
      }
      s_s[row][col] = s;
    }
    __syncthreads();

    // online-softmax row stats
    if (tid < 16) {
      float mold = s_m[tid], mt = mold;
      for (int c = 0; c < TILE_N; ++c) mt = fmaxf(mt, s_s[tid][c]);
      float scale = __expf(mold - mt);
      float lsum = 0.f;
      for (int c = 0; c < TILE_N; ++c) {
        float sv = s_s[tid][c];
        float pv = (sv > -1e29f) ? __expf(sv - mt) : 0.f;
        s_p[tid][c] = (_Float16)pv;
        lsum += pv;
      }
      s_m[tid] = mt;
      s_l[tid] = s_l[tid] * scale + lsum;
      s_scale[tid] = scale;
    }
    __syncthreads();

    // rescale + O += P @ V  (wave owns 64 H-columns -> 4 WMMA tiles)
    float sc[8];
#pragma unroll
    for (int j = 0; j < 8; ++j) sc[j] = s_scale[j + 8 * half];
    v16h ap = cat16(*(const v8h*)(&s_p[m_][half * 8]),
                    *(const v8h*)(&s_p[m_][16 + half * 8]));
#pragma unroll
    for (int c4 = 0; c4 < 4; ++c4) {
#pragma unroll
      for (int j = 0; j < 8; ++j) acc[c4][j] *= sc[j];
      const _Float16* vp = &s_vT[wave * 64 + c4 * 16 + m_][half * 8];
      v16h bv = cat16(*(const v8h*)vp, *(const v8h*)(vp + 16));
      acc[c4] = wmma_f16(ap, bv, acc[c4]);
    }
  }
  __syncthreads();

  // write unnormalized partials
#pragma unroll
  for (int c4 = 0; c4 < 4; ++c4) {
    int hcol = wave * 64 + c4 * 16 + m_;
#pragma unroll
    for (int j = 0; j < 8; ++j) {
      int row = j + 8 * half;
      pO[((size_t)split * RREG + r0 + row) * HDIM + hcol] = acc[c4][j];
    }
  }
  if (tid < 16) {
    pM[split * RREG + r0 + tid] = s_m[tid];
    pL[split * RREG + r0 + tid] = s_l[tid];
  }
}

// ---------------------------------------------------------------------------
// Combine NSPLIT softmax partials -> normalized attn output + has_road flags.
// ---------------------------------------------------------------------------
__global__ __launch_bounds__(256) void combine_kernel(
    const float* __restrict__ pO, const float* __restrict__ pM,
    const float* __restrict__ pL, float* __restrict__ O, int* __restrict__ flags) {
  int idx = blockIdx.x * 256 + threadIdx.x;  // < R*H
  int r = idx >> 9, h = idx & 511;
  float M = -1e30f;
  for (int s = 0; s < NSPLIT; ++s) M = fmaxf(M, pM[s * RREG + r]);
  float L = 0.f, a = 0.f;
  for (int s = 0; s < NSPLIT; ++s) {
    float sc = __expf(pM[s * RREG + r] - M);
    L += pL[s * RREG + r] * sc;
    a += pO[((size_t)s * RREG + r) * HDIM + h] * sc;
  }
  O[idx] = (L > 0.f) ? a / L : 0.f;
  if (h == 0) flags[r] = (L > 0.f) ? 1 : 0;
}

// ---------------------------------------------------------------------------
// res = out @ Wproj^T + region, with empty-region fallback; f16 Wproj.
// ---------------------------------------------------------------------------
__global__ __launch_bounds__(256) void final_proj_kernel(
    const float* __restrict__ O, const _Float16* __restrict__ Wproj,
    const float* __restrict__ region, const int* __restrict__ flags,
    float* __restrict__ res) {
  const int tid = threadIdx.x, wave = tid >> 5, lane = tid & 31;
  const int m_ = lane & 15, half = lane >> 4;
  const int r0 = blockIdx.x * 16;
  const int i0 = blockIdx.y * 128 + wave * 16;
  const float* arow = O + (size_t)(r0 + m_) * HDIM;
  const _Float16* wrow = Wproj + (size_t)(i0 + m_) * HDIM;
  v8f acc0 = {}, acc1 = {};
#pragma unroll 2
  for (int kb = 0; kb < HDIM; kb += 64) {
    v16h a0 = cat16(cvt8(*(const v8f*)(arow + kb + half * 8)),
                    cvt8(*(const v8f*)(arow + kb + 16 + half * 8)));
    acc0 = wmma_f16(a0, load_b_f16(wrow, kb, half), acc0);
    v16h a1 = cat16(cvt8(*(const v8f*)(arow + kb + 32 + half * 8)),
                    cvt8(*(const v8f*)(arow + kb + 48 + half * 8)));
    acc1 = wmma_f16(a1, load_b_f16(wrow, kb + 32, half), acc1);
  }
#pragma unroll
  for (int j = 0; j < 8; ++j) {
    int row = j + 8 * half, col = i0 + m_;
    float rg = region[(size_t)(r0 + row) * HDIM + col];
    res[(size_t)(r0 + row) * HDIM + col] =
        flags[r0 + row] ? (acc0[j] + acc1[j] + rg) : rg;
  }
}

extern "C" void kernel_launch(void* const* d_in, const int* in_sizes, int n_in,
                              void* d_out, int out_size, void* d_ws, size_t ws_size,
                              hipStream_t stream) {
  (void)in_sizes; (void)n_in; (void)out_size; (void)ws_size;
  const float* road_emb = (const float*)d_in[0];
  const int*   assign   = (const int*)d_in[1];
  const float* region   = (const float*)d_in[2];
  const int*   distm    = (const int*)d_in[3];
  const int*   degm     = (const int*)d_in[4];
  const float* Wq       = (const float*)d_in[5];
  const float* Wk       = (const float*)d_in[6];
  const float* Wv       = (const float*)d_in[7];
  const float* Wproj    = (const float*)d_in[8];
  const float* w1       = (const float*)d_in[9];
  const float* w2       = (const float*)d_in[10];
  const float* len_tab  = (const float*)d_in[11];
  const float* deg_tab  = (const float*)d_in[12];

  char* p = (char*)d_ws;
  auto carve = [&](size_t bytes) {
    char* q = p; p += (bytes + 255) & ~(size_t)255; return q;
  };
  _Float16* k_h = (_Float16*)carve((size_t)NROADS * HDIM * 2);
  _Float16* v_h = (_Float16*)carve((size_t)NROADS * HDIM * 2);
  _Float16* q_h = (_Float16*)carve((size_t)RREG * HDIM * 2);
  _Float16* Wq_h = (_Float16*)carve((size_t)HDIM * HDIM * 2);
  _Float16* Wk_h = (_Float16*)carve((size_t)HDIM * HDIM * 2);
  _Float16* Wv_h = (_Float16*)carve((size_t)HDIM * HDIM * 2);
  _Float16* Wp_h = (_Float16*)carve((size_t)HDIM * HDIM * 2);
  float* deg_proj = (float*)carve((size_t)360 * RREG * 4);
  float* len_proj = (float*)carve((size_t)200 * RREG * 4);
  float* pO = (float*)carve((size_t)NSPLIT * RREG * HDIM * 4);
  float* pM = (float*)carve((size_t)NSPLIT * RREG * 4);
  float* pL = (float*)carve((size_t)NSPLIT * RREG * 4);
  float* obuf = (float*)carve((size_t)RREG * HDIM * 4);
  int* flags = (int*)carve((size_t)RREG * 4);

  dim3 blk(256);
  const int wblocks = (HDIM * HDIM / 8) / 256;  // 128 blocks per weight matrix
  cvt_f16_kernel<<<dim3(wblocks), blk, 0, stream>>>(Wq, Wq_h);
  cvt_f16_kernel<<<dim3(wblocks), blk, 0, stream>>>(Wk, Wk_h);
  cvt_f16_kernel<<<dim3(wblocks), blk, 0, stream>>>(Wv, Wv_h);
  cvt_f16_kernel<<<dim3(wblocks), blk, 0, stream>>>(Wproj, Wp_h);
  gemm_kv_kernel<<<dim3(NROADS / 16, HDIM / 256), blk, 0, stream>>>(
      road_emb, Wk_h, Wv_h, k_h, v_h);
  gemm_xwT_f16out<<<dim3(RREG / 16, HDIM / 128), blk, 0, stream>>>(region, Wq_h, q_h);
  table_proj_kernel<<<dim3((560 * RREG) / 256), blk, 0, stream>>>(
      deg_tab, w1, len_tab, w2, deg_proj, len_proj);
  flash_attn_kernel<<<dim3(RREG / 16, NSPLIT), blk, 0, stream>>>(
      q_h, k_h, v_h, deg_proj, len_proj, assign, degm, distm, pO, pM, pL);
  combine_kernel<<<dim3((RREG * HDIM) / 256), blk, 0, stream>>>(pO, pM, pL, obuf, flags);
  final_proj_kernel<<<dim3(RREG / 16, HDIM / 128), blk, 0, stream>>>(
      obuf, Wp_h, region, flags, (float*)d_out);
}
